// SERESkippedQwen3MoeSparseMoeBlock_48593259986979
// MI455X (gfx1250) — matrix operand order repaired and checked by reading, not compiled, not run on hardware
//
#include <hip/hip_runtime.h>

// ---------------- problem constants ----------------
#define T_TOK  2048
#define E_EXP  64
#define TOPK   8
#define SELK   4
#define HDIM   2048
#define FDIM   768
#define F2DIM  1536
#define A_TOT  (T_TOK * TOPK)      // 16384 assignments

// gate_up kernel tiling (M=64, double-buffered async K-chunks)
#define MA     64
#define KCA    128                 // K-chunk staged in LDS (bf16)
#define NCHUNK (HDIM / KCA)        // 16
#define XSTRA  (KCA + 8)           // 136 bf16 -> 272B rows (16B aligned, 4-bank skew)
#define XBUFB  (MA * XSTRA * 2)    // bytes per X buffer
#define ZSPLIT 6                   // 48 gate f-tiles / 8 waves
#define MAXTA  (A_TOT / MA)        // 256

// down kernel tiling (M=32)
#define MB     32
#define HSTRB  (FDIM + 8)          // 776 bf16 -> 1552B rows
#define MAXTB  (A_TOT / MB)        // 512

typedef __attribute__((ext_vector_type(16))) __bf16 v16bf;
typedef __attribute__((ext_vector_type(8)))  __bf16 v8bf;
typedef __attribute__((ext_vector_type(8)))  float  v8f;

__device__ __forceinline__ v16bf frag_cat(v8bf lo, v8bf hi) {
  return __builtin_shufflevector(lo, hi, 0,1,2,3,4,5,6,7,8,9,10,11,12,13,14,15);
}

// A fragment (16x32 bf16, MxK) from LDS row-major.
// ISA 7.12.2: lane m/half: VGPR0-3 K = half*8+0..7, VGPR4-7 K = 16+half*8+0..7
// -> two contiguous 16B LDS reads per lane.
__device__ __forceinline__ v16bf load_a_frag(const __bf16* lds, int stride, int kk, int lane) {
  const int m = lane & 15, half = lane >> 4;
  const __bf16* p = lds + m * stride + kk + (half << 3);
  v8bf lo = *(const v8bf*)(p);
  v8bf hi = *(const v8bf*)(p + 16);
  return frag_cat(lo, hi);
}

// B fragment (32x16 bf16, KxN): B[k][n] = W[nrow0+n][kbase+k], W row-major bf16.
// Lane n/half holds K = half*8 + 0..7 and 16 + half*8 + 0..7 -> 2 x b128 loads.
__device__ __forceinline__ v16bf load_b_frag(const __bf16* __restrict__ W, int rowStride,
                                             int nrow0, int kbase, int lane) {
  const int n = lane & 15, half = lane >> 4;
  const __bf16* p = W + (size_t)(nrow0 + n) * rowStride + kbase + (half << 3);
  v8bf lo = *(const v8bf*)(p);
  v8bf hi = *(const v8bf*)(p + 16);
  return frag_cat(lo, hi);
}

// ---------------- kernel A: f32 -> bf16 bulk convert (8 elems/thread) ----------------
__global__ void moe_cvt_bf16_kernel(const float* __restrict__ src, __bf16* __restrict__ dst, int n8) {
  int i = blockIdx.x * blockDim.x + threadIdx.x;
  if (i >= n8) return;
  const float4* s4 = (const float4*)(src) + (size_t)i * 2;
  float4 a = s4[0], b = s4[1];
  v8bf o;
  o[0] = (__bf16)a.x; o[1] = (__bf16)a.y; o[2] = (__bf16)a.z; o[3] = (__bf16)a.w;
  o[4] = (__bf16)b.x; o[5] = (__bf16)b.y; o[6] = (__bf16)b.z; o[7] = (__bf16)b.w;
  *((v8bf*)dst + i) = o;
}

// ---------------- kernel 0: zero output + control ----------------
__global__ void moe_zero_kernel(float* __restrict__ out, int n, int* __restrict__ ctrl) {
  int idx = blockIdx.x * blockDim.x + threadIdx.x;
  if (idx < n) out[idx] = 0.f;
  if (idx < 4 * E_EXP) ctrl[idx] = 0;   // primary | secondary | counts | cursor
}

// ---------------- kernel 1: router ----------------
__global__ void moe_router_kernel(const float* __restrict__ x, const float* __restrict__ gw,
                                  int* __restrict__ top_idx, float* __restrict__ top_val,
                                  int* __restrict__ primary, int* __restrict__ secondary) {
  __shared__ float probs[E_EXP];
  const int t = blockIdx.x;
  const int e = threadIdx.x;          // blockDim.x == 64
  const float4* x4 = (const float4*)(x + (size_t)t * HDIM);
  const float4* g4 = (const float4*)(gw + (size_t)e * HDIM);
  float acc = 0.f;
  for (int i = 0; i < HDIM / 4; ++i) {
    float4 a = x4[i], b = g4[i];
    acc += a.x * b.x + a.y * b.y + a.z * b.z + a.w * b.w;
  }
  probs[e] = acc;
  __syncthreads();
  if (e == 0) {
    float mx = -1e30f;
    for (int i = 0; i < E_EXP; ++i) mx = probs[i] > mx ? probs[i] : mx;
    for (int i = 0; i < E_EXP; ++i) probs[i] = __expf(probs[i] - mx);
    float tv[TOPK]; int ti[TOPK];
    for (int k = 0; k < TOPK; ++k) {
      float bv = -1.f; int bi = 0;
      for (int i = 0; i < E_EXP; ++i) if (probs[i] > bv) { bv = probs[i]; bi = i; }
      tv[k] = bv; ti[k] = bi; probs[bi] = -2.f;
    }
    float s = 0.f;
    for (int k = 0; k < TOPK; ++k) s += tv[k];
    if (s < 1e-12f) s = 1e-12f;
    for (int k = 0; k < TOPK; ++k) {
      top_idx[t * TOPK + k] = ti[k];
      top_val[t * TOPK + k] = tv[k] / s;   // exp-ratio == prob-ratio
      if (k < SELK) atomicOr(&primary[ti[k]], 1);
      else          atomicOr(&secondary[ti[k]], 1);
    }
  }
}

// ---------------- kernel 2: SERE similarity mapping ----------------
__global__ void moe_sere_kernel(const float* __restrict__ sim, const int* __restrict__ primary,
                                const int* __restrict__ secondary, int* __restrict__ mapping) {
  const int e = threadIdx.x;          // 64 threads
  float best = -1e30f; int bp = 0; bool any = false;
  for (int p = 0; p < E_EXP; ++p) {
    if (primary[p]) {
      float v = (p == e) ? 1.0f : sim[e * E_EXP + p];
      if (!any || v > best) { best = v; bp = p; any = true; }
    }
  }
  int m = e;
  if (secondary[e] && !primary[e] && any && best >= 0.5f) m = bp;
  mapping[e] = m;
}

// ---------------- kernel 3: reroute + counts ----------------
__global__ void moe_reroute_kernel(const int* __restrict__ top_idx, const int* __restrict__ mapping,
                                   int* __restrict__ rerouted, int* __restrict__ counts) {
  int a = blockIdx.x * blockDim.x + threadIdx.x;
  if (a >= A_TOT) return;
  int slot = a & (TOPK - 1);
  int e = top_idx[a];
  int re = (slot < SELK) ? e : mapping[e];
  rerouted[a] = re;
  atomicAdd(&counts[re], 1);
}

// ---------------- kernel 4: exclusive scan ----------------
__global__ void moe_scan_kernel(const int* __restrict__ counts, int* __restrict__ offsets) {
  if (threadIdx.x == 0) {
    int s = 0;
    for (int e = 0; e < E_EXP; ++e) { offsets[e] = s; s += counts[e]; }
    offsets[E_EXP] = s;
  }
}

// ---------------- kernel 5: scatter into expert-sorted order ----------------
__global__ void moe_scatter_kernel(const int* __restrict__ rerouted, const float* __restrict__ top_val,
                                   const int* __restrict__ offsets, int* __restrict__ cursor,
                                   int* __restrict__ asg_tok, float* __restrict__ asg_w) {
  int a = blockIdx.x * blockDim.x + threadIdx.x;
  if (a >= A_TOT) return;
  int e = rerouted[a];
  int pos = offsets[e] + atomicAdd(&cursor[e], 1);
  asg_tok[pos] = a >> 3;
  asg_w[pos]  = top_val[a];
}

// ---------------- kernel 6: gate_up + SwiGLU -> bf16 hact ----------------
// M=64, N split 6-way (1 gate/up f-tile pair per wave), double-buffered
// async-to-LDS X staging hidden behind WMMA.
__launch_bounds__(256)
__global__ void moe_gateup_kernel(const __bf16* __restrict__ xbf,
                                  const __bf16* __restrict__ gup,
                                  const int* __restrict__ offsets,
                                  const int* __restrict__ counts,
                                  const int* __restrict__ asg_tok,
                                  __bf16* __restrict__ hact) {
  __shared__ __bf16 Xs[2][MA * XSTRA];   // 34,816 B double buffer
  __shared__ int    tokS[MA];

  const int e    = blockIdx.z;
  const int z    = blockIdx.y;        // 0..5
  const int tile = blockIdx.x;
  const int cnt  = counts[e];
  if (tile * MA >= cnt) return;       // uniform -> EXEC all-ones at WMMA
  const int start = offsets[e] + tile * MA;
  const int rem   = cnt - tile * MA;
  const int nrow  = rem < MA ? rem : MA;

  const int tid = threadIdx.x;
  if (tid < MA) tokS[tid] = (tid < nrow) ? asg_tok[start + tid] : 0;
  __syncthreads();                    // tokS visible before async issue

  const int wave = tid >> 5, lane = tid & 31;
  const __bf16* gupE = gup + (size_t)e * F2DIM * HDIM;   // [1536][2048] bf16
  const int ft = z * 8 + wave;        // this wave's gate f-tile (0..47)
  const unsigned XsBase = (unsigned)(uintptr_t)(&Xs[0][0]);

  // stage one K-chunk into buffer `buf` (raw bf16 copy, 4 x 16B per thread)
  auto stage = [&](int buf, int kc) {
    for (int c = tid; c < MA * (KCA / 8); c += 256) {   // 1024 chunks, uniform
      int r = c >> 4, k8 = c & 15;
      const __bf16* gp = xbf + (size_t)tokS[r] * HDIM + kc * KCA + k8 * 8;
      unsigned loff = XsBase + buf * XBUFB + r * (XSTRA * 2) + k8 * 16;
      asm volatile("global_load_async_to_lds_b128 %0, %1, off"
                   :: "v"(loff), "v"(gp) : "memory");
    }
  };

  v8f zero = {};
  v8f accG[4], accU[4];
#pragma unroll
  for (int m = 0; m < 4; ++m) { accG[m] = zero; accU[m] = zero; }

  stage(0, 0);
  for (int kc = 0; kc < NCHUNK; ++kc) {
    asm volatile("s_wait_asynccnt 0x0" ::: "memory");
    __syncthreads();                  // all waves' loads for `cur` landed; prev compute done
    const int cur = kc & 1;
    if (kc + 1 < NCHUNK) stage(cur ^ 1, kc + 1);   // overlap next chunk with compute
    const __bf16* Xc = &Xs[cur][0];
#pragma unroll
    for (int kk = 0; kk < KCA; kk += 32) {
      v16bf a0 = load_a_frag(Xc + 0 * 16 * XSTRA, XSTRA, kk, lane);
      v16bf a1 = load_a_frag(Xc + 1 * 16 * XSTRA, XSTRA, kk, lane);
      v16bf a2 = load_a_frag(Xc + 2 * 16 * XSTRA, XSTRA, kk, lane);
      v16bf a3 = load_a_frag(Xc + 3 * 16 * XSTRA, XSTRA, kk, lane);
      v16bf bg = load_b_frag(gupE, HDIM, ft * 16, kc * KCA + kk, lane);
      accG[0] = __builtin_amdgcn_wmma_f32_16x16x32_bf16(false, a0, false, bg, (short)0, accG[0], false, false);
      accG[1] = __builtin_amdgcn_wmma_f32_16x16x32_bf16(false, a1, false, bg, (short)0, accG[1], false, false);
      accG[2] = __builtin_amdgcn_wmma_f32_16x16x32_bf16(false, a2, false, bg, (short)0, accG[2], false, false);
      accG[3] = __builtin_amdgcn_wmma_f32_16x16x32_bf16(false, a3, false, bg, (short)0, accG[3], false, false);
      v16bf bu = load_b_frag(gupE, HDIM, FDIM + ft * 16, kc * KCA + kk, lane);
      accU[0] = __builtin_amdgcn_wmma_f32_16x16x32_bf16(false, a0, false, bu, (short)0, accU[0], false, false);
      accU[1] = __builtin_amdgcn_wmma_f32_16x16x32_bf16(false, a1, false, bu, (short)0, accU[1], false, false);
      accU[2] = __builtin_amdgcn_wmma_f32_16x16x32_bf16(false, a2, false, bu, (short)0, accU[2], false, false);
      accU[3] = __builtin_amdgcn_wmma_f32_16x16x32_bf16(false, a3, false, bu, (short)0, accU[3], false, false);
    }
  }

  // SwiGLU epilogue: fast sigmoid via v_rcp_f32; guard rows beyond expert range.
  const int col = lane & 15, half = lane >> 4;
#pragma unroll
  for (int m = 0; m < 4; ++m) {
#pragma unroll
    for (int r = 0; r < 8; ++r) {
      int M = m * 16 + r + half * 8;  // C/D layout: VGPR r -> M = r + 8*half
      if (M < nrow) {
        float g = accG[m][r];
        float sig = __builtin_amdgcn_rcpf(1.f + __expf(-g));
        hact[(size_t)(start + M) * FDIM + ft * 16 + col] = (__bf16)(g * sig * accU[m][r]);
      }
    }
  }
}

// ---------------- kernel 7: down projection + weighted combine (M=32) ----------------
__launch_bounds__(256)
__global__ void moe_down_kernel(const __bf16* __restrict__ hact,
                                const __bf16* __restrict__ dwn,
                                const int* __restrict__ offsets,
                                const int* __restrict__ counts,
                                const int* __restrict__ asg_tok,
                                const float* __restrict__ asg_w,
                                float* __restrict__ out) {
  __shared__ __bf16 Hs[MB * HSTRB];   // 49,664 B
  __shared__ int    tokS[MB];
  __shared__ float  wS[MB];

  const int e    = blockIdx.y;
  const int tile = blockIdx.x;
  const int cnt  = counts[e];
  if (tile * MB >= cnt) return;
  const int start = offsets[e] + tile * MB;
  const int rem   = cnt - tile * MB;
  const int nrow  = rem < MB ? rem : MB;

  const int tid = threadIdx.x;
  // async stage hact tile first (raw bf16 copy: 32 rows x 96 x 16B)
  {
    unsigned ldsBase = (unsigned)(uintptr_t)(&Hs[0]);
    for (int c = tid; c < MB * (FDIM / 8); c += 256) {   // 12 iters, uniform
      int r = c / 96, k8 = c % 96;
      const __bf16* gp = hact + (size_t)(start + r) * FDIM + k8 * 8;
      unsigned loff = ldsBase + r * (HSTRB * 2) + k8 * 16;
      asm volatile("global_load_async_to_lds_b128 %0, %1, off"
                   :: "v"(loff), "v"(gp) : "memory");
    }
  }
  if (tid < MB) {
    tokS[tid] = (tid < nrow) ? asg_tok[start + tid] : 0;
    wS[tid]   = (tid < nrow) ? asg_w[start + tid] : 0.f;   // rows past nrow contribute 0
  }
  asm volatile("s_wait_asynccnt 0x0" ::: "memory");
  __syncthreads();

  const int wave = tid >> 5, lane = tid & 31;
  const __bf16* dwnE = dwn + (size_t)e * HDIM * FDIM;    // [2048][768] bf16
  const int col = lane & 15, half = lane >> 4;
  v8f zero = {};

  for (int i = 0; i < 16; ++i) {
    const int nt = wave * 16 + i;     // 128 N-tiles over H=2048
    v8f acc0 = zero, acc1 = zero;
    for (int kk = 0; kk < FDIM; kk += 32) {
      v16bf b  = load_b_frag(dwnE, FDIM, nt * 16, kk, lane);
      v16bf a0 = load_a_frag(Hs,              HSTRB, kk, lane);
      v16bf a1 = load_a_frag(Hs + 16 * HSTRB, HSTRB, kk, lane);
      acc0 = __builtin_amdgcn_wmma_f32_16x16x32_bf16(false, a0, false, b, (short)0, acc0, false, false);
      acc1 = __builtin_amdgcn_wmma_f32_16x16x32_bf16(false, a1, false, b, (short)0, acc1, false, false);
    }
    const int h = nt * 16 + col;
#pragma unroll
    for (int r = 0; r < 8; ++r) {
      int M0 = r + half * 8, M1 = 16 + M0;
      atomicAdd(&out[(size_t)tokS[M0] * HDIM + h], acc0[r] * wS[M0]);
      atomicAdd(&out[(size_t)tokS[M1] * HDIM + h], acc1[r] * wS[M1]);
    }
  }
}

// ---------------- launch ----------------
extern "C" void kernel_launch(void* const* d_in, const int* in_sizes, int n_in,
                              void* d_out, int out_size, void* d_ws, size_t ws_size,
                              hipStream_t stream) {
  (void)in_sizes; (void)n_in; (void)out_size; (void)ws_size;
  const float* x   = (const float*)d_in[0];   // [1,2048,2048]
  const float* gw  = (const float*)d_in[1];   // [64,2048]
  const float* gup = (const float*)d_in[2];   // [64,1536,2048]
  const float* dwn = (const float*)d_in[3];   // [64,2048,768]
  const float* sim = (const float*)d_in[4];   // [64,64]
  float* out = (float*)d_out;

  // ---- workspace layout ----
  int* ws_i = (int*)d_ws;
  int*    top_idx   = ws_i;                          // A_TOT ints
  float*  top_val   = (float*)(ws_i + A_TOT);        // A_TOT floats
  int*    rerouted  = ws_i + 2 * A_TOT;              // A_TOT ints
  int*    asg_tok   = ws_i + 3 * A_TOT;              // A_TOT ints
  float*  asg_w     = (float*)(ws_i + 4 * A_TOT);    // A_TOT floats
  int*    ctrl      = ws_i + 5 * A_TOT;
  int*    primary   = ctrl;                          // 64
  int*    secondary = ctrl + E_EXP;                  // 64
  int*    counts    = ctrl + 2 * E_EXP;              // 64
  int*    cursor    = ctrl + 3 * E_EXP;              // 64
  int*    mapping   = ctrl + 4 * E_EXP;              // 64
  int*    offsets   = ctrl + 5 * E_EXP;              // 65

  char*  wsb = (char*)d_ws;
  size_t o   = (size_t)6 * A_TOT * sizeof(int);
  __bf16* hact  = (__bf16*)(wsb + o); o += (size_t)(A_TOT + 64) * FDIM * 2;      // 25.3 MB
  __bf16* xbf   = (__bf16*)(wsb + o); o += (size_t)T_TOK * HDIM * 2;             //  8.4 MB
  __bf16* gupbf = (__bf16*)(wsb + o); o += (size_t)E_EXP * F2DIM * HDIM * 2;     //  403 MB
  __bf16* dwnbf = (__bf16*)(wsb + o);                                            //  201 MB

  // ---- one-time bf16 conversions (weights reused ~16x from L2 per call) ----
  {
    int n8;
    n8 = T_TOK * HDIM / 8;
    moe_cvt_bf16_kernel<<<(n8 + 255) / 256, 256, 0, stream>>>(x, xbf, n8);
    n8 = (int)((size_t)E_EXP * F2DIM * HDIM / 8);
    moe_cvt_bf16_kernel<<<(n8 + 255) / 256, 256, 0, stream>>>(gup, gupbf, n8);
    n8 = (int)((size_t)E_EXP * HDIM * FDIM / 8);
    moe_cvt_bf16_kernel<<<(n8 + 255) / 256, 256, 0, stream>>>(dwn, dwnbf, n8);
  }

  const int nOut = T_TOK * HDIM;
  moe_zero_kernel<<<(nOut + 255) / 256, 256, 0, stream>>>(out, nOut, ctrl);
  moe_router_kernel<<<T_TOK, E_EXP, 0, stream>>>(x, gw, top_idx, top_val, primary, secondary);
  moe_sere_kernel<<<1, E_EXP, 0, stream>>>(sim, primary, secondary, mapping);
  moe_reroute_kernel<<<A_TOT / 256, 256, 0, stream>>>(top_idx, mapping, rerouted, counts);
  moe_scan_kernel<<<1, 32, 0, stream>>>(counts, offsets);
  moe_scatter_kernel<<<A_TOT / 256, 256, 0, stream>>>(rerouted, top_val, offsets, cursor,
                                                      asg_tok, asg_w);
  dim3 gridA(MAXTA, ZSPLIT, E_EXP);
  moe_gateup_kernel<<<gridA, 256, 0, stream>>>(xbf, gupbf, offsets, counts, asg_tok, hact);
  dim3 gridB(MAXTB, E_EXP);
  moe_down_kernel<<<gridB, 256, 0, stream>>>(hact, dwnbf, offsets, counts, asg_tok, asg_w, out);
}
